// MultiScaleAttention_87711822119334
// MI455X (gfx1250) — matrix operand order repaired and checked by reading, not compiled, not run on hardware
//
#include <hip/hip_runtime.h>
#include <hip/hip_bf16.h>
#include <math.h>

// ---------------- problem constants ----------------
#define BB      2
#define SS      2048
#define DD      1024
#define HH      16
#define DKK     64
#define LWW     256
#define GST     64
#define GGC     (SS / GST)     // 32 global tokens
#define NWW     (SS / LWW)     // 8 local windows
#define SWHALF  256
#define EPSV    1e-5f
#define SCALEV  0.125f         // 1/sqrt(64)
#define NEGBIG  (-1.0e30f)

typedef __attribute__((ext_vector_type(16))) __bf16 v16bf;
typedef __attribute__((ext_vector_type(8)))  float  v8f;

// fp32 -> bf16 round-to-nearest-even (scalar)
__device__ __forceinline__ __bf16 f2bf(float f) {
    union { float f; unsigned u; } a; a.f = f;
    unsigned r = a.u + 0x7FFFu + ((a.u >> 16) & 1u);
    union { unsigned short s; __bf16 b; } c; c.s = (unsigned short)(r >> 16);
    return c.b;
}

// pack two fp32 -> two bf16 (RNE) in one dword
#if __has_builtin(__builtin_amdgcn_cvt_pk_bf16_f32)
typedef __attribute__((ext_vector_type(2))) __bf16 v2bf;
__device__ __forceinline__ unsigned pack2bf(float x, float y) {
    union { v2bf v; unsigned u; } r;
    r.v = __builtin_amdgcn_cvt_pk_bf16_f32(x, y);
    return r.u;
}
#else
__device__ __forceinline__ unsigned pack2bf(float x, float y) {
    union { float f; unsigned u; } a, b; a.f = x; b.f = y;
    unsigned lo = (a.u + 0x7FFFu + ((a.u >> 16) & 1u)) >> 16;
    unsigned hi = (b.u + 0x7FFFu + ((b.u >> 16) & 1u)) & 0xFFFF0000u;
    return lo | hi;
}
#endif

// A fragment (16x32 MxK): lane (l15,lh) holds row m=l15, K = lh*8+2j and 16+lh*8+2j
// 'Arow' points at a contiguous 32-element k-window of the row.
__device__ __forceinline__ v16bf load_a_frag(const __bf16* Arow, int kb) {
    union { v16bf v; unsigned u[8]; } r;
    const unsigned* p = (const unsigned*)Arow;
#pragma unroll
    for (int j = 0; j < 4; ++j) r.u[j]     = p[(kb >> 1) + j];
#pragma unroll
    for (int j = 0; j < 4; ++j) r.u[4 + j] = p[8 + (kb >> 1) + j];
    return r.v;
}

// B fragment (32x16 KxN): lane (l15,lh) holds col n=l15, K = lh*8+2j and 16+lh*8+2j, j=0..3
// 'Brow' points at a contiguous 32-element k-window of column n's data.
__device__ __forceinline__ v16bf load_b_frag(const __bf16* Brow, int kb2) {
    union { v16bf v; unsigned u[8]; } r;
    const unsigned* p = (const unsigned*)Brow;
#pragma unroll
    for (int j = 0; j < 8; ++j) r.u[j] = p[(j < 4 ? 0 : 4) + (kb2 >> 2) + j];
    return r.v;
}

#define WMMA_BF16(A, B, C) \
    __builtin_amdgcn_wmma_f32_16x16x32_bf16(false, (A), false, (B), (short)0, (C), false, false)

// ---------------- WMMA GEMM:  C[M,N] = A[M,K] @ W[K,N] + bias ----------------
// Block tile 128(M) x 128(N), K-step 32; 256 threads = 8 waves (4 M-sub x 2 N-sub),
// each wave owns a 32x64 tile = 8 v_wmma_f32_16x16x32_bf16 accumulators.
__global__ __launch_bounds__(256) void gemm_bias_wmma(
    const float* __restrict__ A, int lda, int M, int K,
    const float* __restrict__ W, int N,
    const float* __restrict__ bias,
    float* __restrict__ C, int ldc, int coloff)
{
    __shared__ __bf16 As[128][32];    // 8 KB
    __shared__ __bf16 BsT[128][32];   // 8 KB

    const int tid  = threadIdx.x;
    const int n0   = blockIdx.x * 128;
    const int m0   = blockIdx.y * 128;
    const int lane = tid & 31;
    const int wv   = tid >> 5;
    const int wm   = wv & 3;           // 4 M sub-tiles of 32 rows
    const int wn   = wv >> 2;          // 2 N sub-tiles of 64 cols
    const int l15  = lane & 15;
    const int lh   = lane >> 4;

    // staging ownership
    const int amm  = tid >> 1;             // A: row within tile
    const int akk  = (tid & 1) * 16;       // A: 16-element k-half
    int arow = m0 + amm; arow = (arow < M) ? arow : (M - 1);
    const int bnn  = tid & 127;            // B: column within tile
    const int bks  = (tid >> 7) * 16;      // B: 16-element k-segment

    v8f acc[2][4];
#pragma unroll
    for (int i = 0; i < 2; ++i)
#pragma unroll
        for (int j = 0; j < 4; ++j)
#pragma unroll
            for (int r = 0; r < 8; ++r) acc[i][j][r] = 0.0f;

    for (int k0 = 0; k0 < K; k0 += 32) {
        __syncthreads();
        // Stage A tile (128x32) bf16: 16 contiguous k per thread, packed dword stores.
        {
            const float* Ap = &A[(size_t)arow * lda + k0 + akk];
            unsigned* dst = (unsigned*)&As[amm][akk];
            unsigned pk[8];
#pragma unroll
            for (int j = 0; j < 8; ++j) pk[j] = pack2bf(Ap[2 * j], Ap[2 * j + 1]);
#pragma unroll
            for (int j = 0; j < 8; ++j) dst[j] = pk[j];
        }
        // Stage W tile transposed (BsT[n][k]): 16 k per thread down column n.
        {
            unsigned* dst = (unsigned*)&BsT[bnn][bks];
            unsigned pk[8];
#pragma unroll
            for (int j = 0; j < 8; ++j) {
                float f0 = W[(size_t)(k0 + bks + 2 * j) * N + n0 + bnn];
                float f1 = W[(size_t)(k0 + bks + 2 * j + 1) * N + n0 + bnn];
                pk[j] = pack2bf(f0, f1);
            }
#pragma unroll
            for (int j = 0; j < 8; ++j) dst[j] = pk[j];
        }
        // Prefetch next K-tile (global_prefetch_b8).
        if (k0 + 32 < K) {
            __builtin_prefetch(&W[(size_t)(k0 + 32 + bks) * N + n0 + bnn], 0, 0);
            __builtin_prefetch(&A[(size_t)arow * lda + k0 + 32 + akk], 0, 0);
        }
        __syncthreads();

        v16bf a0 = load_a_frag(&As[wm * 32 +      l15][0], lh * 8);
        v16bf a1 = load_a_frag(&As[wm * 32 + 16 + l15][0], lh * 8);
#pragma unroll
        for (int ns = 0; ns < 4; ++ns) {
            v16bf b = load_b_frag(&BsT[wn * 64 + ns * 16 + l15][0], lh * 16);
            acc[0][ns] = WMMA_BF16(a0, b, acc[0][ns]);
            acc[1][ns] = WMMA_BF16(a1, b, acc[1][ns]);
        }
    }

#pragma unroll
    for (int ms = 0; ms < 2; ++ms)
#pragma unroll
        for (int ns = 0; ns < 4; ++ns) {
            int col = n0 + wn * 64 + ns * 16 + l15;
            float bv = bias[col];
#pragma unroll
            for (int r = 0; r < 8; ++r) {
                int row = m0 + wm * 32 + ms * 16 + lh * 8 + r;
                if (row < M) C[(size_t)row * ldc + coloff + col] = acc[ms][ns][r] + bv;
            }
        }
}

// ---------------- WMMA flash-attention core ----------------
// Block: 256 threads = 8 waves; 256 query rows (32 per wave); keys in chunks of 32.
// S = Q K^T via WMMA (inner dim dk=64 -> 2 steps), online softmax in C-layout,
// P bounced through per-wave LDS tile to A-fragment layout, O += P V via WMMA.
template<bool BAND>
__device__ __forceinline__ void attn_core(
    const float* __restrict__ Qp,   // &Q[qrow0*DD + dk_off]
    const float* __restrict__ Kp,   // &K[krow0*DD + dk_off]
    const float* __restrict__ Vp,
    float* __restrict__ Op,         // &Out[qrow0*ld_out + coloff + dk_off]
    int ld_out, int nkeys, int qidx0, int kidx0)
{
    __shared__ __bf16 Qs[256][64];   // 32 KB
    __shared__ __bf16 Ks[32][64];    //  4 KB  (row = key, col = dk)
    __shared__ __bf16 VsT[64][32];   //  4 KB  (row = dk,  col = key)
    __shared__ __bf16 Ps[8][32][32]; // 16 KB  (per-wave P tile)

    const int tid  = threadIdx.x;
    const int lane = tid & 31;
    const int wv   = tid >> 5;
    const int l15  = lane & 15;
    const int lh   = lane >> 4;
    const int qb   = wv * 32;

    // stage Q once: each thread packs one full 64-dk row.
    {
        const float* qrow = &Qp[(size_t)tid * DD];
        unsigned* dst = (unsigned*)&Qs[tid][0];
        unsigned pk[32];
#pragma unroll
        for (int j = 0; j < 32; ++j) pk[j] = pack2bf(qrow[2 * j], qrow[2 * j + 1]);
#pragma unroll
        for (int j = 0; j < 32; ++j) dst[j] = pk[j];
    }
    __syncthreads();

    v16bf a_q[2][2];
#pragma unroll
    for (int ms = 0; ms < 2; ++ms)
#pragma unroll
        for (int c = 0; c < 2; ++c)
            a_q[ms][c] = load_a_frag(&Qs[qb + ms * 16 + l15][c * 32], lh * 8);

    float mx[2][8], ls[2][8];
    v8f acc_o[2][4];
#pragma unroll
    for (int ms = 0; ms < 2; ++ms)
#pragma unroll
        for (int r = 0; r < 8; ++r) { mx[ms][r] = NEGBIG; ls[ms][r] = 0.0f; }
#pragma unroll
    for (int ms = 0; ms < 2; ++ms)
#pragma unroll
        for (int nt = 0; nt < 4; ++nt)
#pragma unroll
            for (int r = 0; r < 8; ++r) acc_o[ms][nt][r] = 0.0f;

    for (int kc = 0; kc < nkeys; kc += 32) {
        __syncthreads();
        // stage K chunk rows (packed) and V chunk transposed (scalar b16).
        {
            int kr = tid >> 3, cc0 = (tid & 7) * 8;   // 32 rows x 8x8 cols
            const float* krow = &Kp[(size_t)(kc + kr) * DD + cc0];
            const float* vrow = &Vp[(size_t)(kc + kr) * DD + cc0];
            unsigned* dst = (unsigned*)&Ks[kr][cc0];
            unsigned pk[4];
#pragma unroll
            for (int j = 0; j < 4; ++j) pk[j] = pack2bf(krow[2 * j], krow[2 * j + 1]);
#pragma unroll
            for (int j = 0; j < 4; ++j) dst[j] = pk[j];
#pragma unroll
            for (int j = 0; j < 8; ++j) VsT[cc0 + j][kr] = f2bf(vrow[j]);
        }
        __syncthreads();

        bool active = true;
        if (BAND) {
            int qlo = qidx0 + qb, qhi = qlo + 31;
            int klo = kidx0 + kc, khi = klo + 31;
            active = (klo <= qhi + SWHALF) && (khi >= qlo - SWHALF);
        }
        if (active) {
            // S = Q K^T  (32x32 per wave)
            v8f acc_s[2][2];
#pragma unroll
            for (int i = 0; i < 2; ++i)
#pragma unroll
                for (int j = 0; j < 2; ++j)
#pragma unroll
                    for (int r = 0; r < 8; ++r) acc_s[i][j][r] = 0.0f;
#pragma unroll
            for (int c = 0; c < 2; ++c) {
                v16bf b0 = load_b_frag(&Ks[l15][c * 32], lh * 16);
                v16bf b1 = load_b_frag(&Ks[16 + l15][c * 32], lh * 16);
                acc_s[0][0] = WMMA_BF16(a_q[0][c], b0, acc_s[0][0]);
                acc_s[0][1] = WMMA_BF16(a_q[0][c], b1, acc_s[0][1]);
                acc_s[1][0] = WMMA_BF16(a_q[1][c], b0, acc_s[1][0]);
                acc_s[1][1] = WMMA_BF16(a_q[1][c], b1, acc_s[1][1]);
            }
            // online softmax (C-layout: row = lh*8+r, col = l15 / 16+l15)
#pragma unroll
            for (int ms = 0; ms < 2; ++ms) {
#pragma unroll
                for (int r = 0; r < 8; ++r) {
                    float s0 = acc_s[ms][0][r] * SCALEV;
                    float s1 = acc_s[ms][1][r] * SCALEV;
                    if (BAND) {
                        int q  = qidx0 + qb + ms * 16 + lh * 8 + r;
                        int k0 = kidx0 + kc + l15;
                        int d0 = q - k0;        d0 = d0 < 0 ? -d0 : d0;
                        int d1 = q - (k0 + 16); d1 = d1 < 0 ? -d1 : d1;
                        if (d0 > SWHALF) s0 = NEGBIG;
                        if (d1 > SWHALF) s1 = NEGBIG;
                    }
                    float cmax = fmaxf(s0, s1);
#pragma unroll
                    for (int m = 1; m <= 8; m <<= 1) cmax = fmaxf(cmax, __shfl_xor(cmax, m, 32));
                    float mnew = fmaxf(mx[ms][r], cmax);
                    float corr = __expf(mx[ms][r] - mnew);
                    float p0 = __expf(s0 - mnew);
                    float p1 = __expf(s1 - mnew);
                    if (BAND) {
                        if (s0 < -1e29f) p0 = 0.0f;
                        if (s1 < -1e29f) p1 = 0.0f;
                    }
                    float psum = p0 + p1;
#pragma unroll
                    for (int m = 1; m <= 8; m <<= 1) psum += __shfl_xor(psum, m, 32);
                    ls[ms][r] = ls[ms][r] * corr + psum;
                    mx[ms][r] = mnew;
                    int prow = ms * 16 + lh * 8 + r;
                    Ps[wv][prow][l15]      = f2bf(p0);
                    Ps[wv][prow][16 + l15] = f2bf(p1);
#pragma unroll
                    for (int nt = 0; nt < 4; ++nt) acc_o[ms][nt][r] *= corr;
                }
            }
            // compiler barrier: keep Ps fragment loads after the bf16 stores
            // (same-wave LDS ops are in-order in hardware).
            asm volatile("" ::: "memory");
            // O += P V
#pragma unroll
            for (int ms = 0; ms < 2; ++ms) {
                v16bf a_p = load_a_frag(&Ps[wv][ms * 16 + l15][0], lh * 8);
#pragma unroll
                for (int nt = 0; nt < 4; ++nt) {
                    v16bf b_v = load_b_frag(&VsT[nt * 16 + l15][0], lh * 16);
                    acc_o[ms][nt] = WMMA_BF16(a_p, b_v, acc_o[ms][nt]);
                }
            }
        }
    }

    // epilogue: normalize and store
#pragma unroll
    for (int ms = 0; ms < 2; ++ms)
#pragma unroll
        for (int r = 0; r < 8; ++r) {
            float inv = 1.0f / ls[ms][r];
            int row = qb + ms * 16 + lh * 8 + r;
#pragma unroll
            for (int nt = 0; nt < 4; ++nt)
                Op[(size_t)row * ld_out + nt * 16 + l15] = acc_o[ms][nt][r] * inv;
        }
}

// ---------------- attention wrappers ----------------
__global__ __launch_bounds__(256) void local_attn_wmma(
    const float* __restrict__ Q, const float* __restrict__ K, const float* __restrict__ V,
    float* __restrict__ Out, int ld_out, int coloff)
{
    int blk = blockIdx.x;
    int h = blk % HH; int w = (blk / HH) % NWW; int b = blk / (HH * NWW);
    size_t rowbase = (size_t)b * SS + (size_t)w * LWW;
    attn_core<false>(Q + rowbase * DD + h * DKK,
                     K + rowbase * DD + h * DKK,
                     V + rowbase * DD + h * DKK,
                     Out + rowbase * ld_out + coloff + h * DKK,
                     ld_out, LWW, 0, 0);
}

__global__ __launch_bounds__(256) void global_attn_wmma(
    const float* __restrict__ Q, const float* __restrict__ K, const float* __restrict__ V,
    float* __restrict__ Out, int ld_out, int coloff)
{
    int blk = blockIdx.x;
    int nqb = SS / 256;
    int qb = blk % nqb; int bh = blk / nqb; int h = bh % HH; int b = bh / HH;
    size_t qbase = (size_t)b * SS + (size_t)qb * 256;
    size_t kbase = (size_t)b * GGC;
    attn_core<false>(Q + qbase * DD + h * DKK,
                     K + kbase * DD + h * DKK,
                     V + kbase * DD + h * DKK,
                     Out + qbase * ld_out + coloff + h * DKK,
                     ld_out, GGC, 0, 0);
}

__global__ __launch_bounds__(256) void sliding_attn_wmma(
    const float* __restrict__ Q, const float* __restrict__ K, const float* __restrict__ V,
    float* __restrict__ Out)
{
    int blk = blockIdx.x;
    int nqb = SS / 256;
    int qb = blk % nqb; int bh = blk / nqb; int h = bh % HH; int b = bh / HH;
    int q0 = qb * 256;
    int kstart = q0 - SWHALF; if (kstart < 0) kstart = 0;
    int kend   = q0 + 256 + SWHALF; if (kend > SS) kend = SS;
    size_t qbase = (size_t)b * SS + q0;
    size_t kbase = (size_t)b * SS + kstart;
    attn_core<true>(Q + qbase * DD + h * DKK,
                    K + kbase * DD + h * DKK,
                    V + kbase * DD + h * DKK,
                    Out + qbase * DD + h * DKK,
                    DD, kend - kstart, q0, kstart);
}

// ---------------- fused residual-add + LayerNorm ----------------
__global__ __launch_bounds__(256) void add_ln_kernel(
    const float* __restrict__ X, const float* __restrict__ Y,
    const float* __restrict__ G, const float* __restrict__ Bt,
    float* __restrict__ Out, int ld_out, int coloff)
{
    __shared__ float rs[256], rs2[256];
    int row = blockIdx.x, tid = threadIdx.x;
    const float* x = X + (size_t)row * DD;
    const float* y = Y + (size_t)row * DD;
    float v[4], s = 0.0f, s2 = 0.0f;
#pragma unroll
    for (int i = 0; i < 4; ++i) {
        float t = x[tid * 4 + i] + y[tid * 4 + i];
        v[i] = t; s += t; s2 += t * t;
    }
    rs[tid] = s; rs2[tid] = s2;
    __syncthreads();
    for (int off = 128; off > 0; off >>= 1) {
        if (tid < off) { rs[tid] += rs[tid + off]; rs2[tid] += rs2[tid + off]; }
        __syncthreads();
    }
    float mean = rs[0] * (1.0f / DD);
    float var  = rs2[0] * (1.0f / DD) - mean * mean;
    float rstd = rsqrtf(var + EPSV);
    float* o = Out + (size_t)row * ld_out + coloff;
#pragma unroll
    for (int i = 0; i < 4; ++i) {
        int c = tid * 4 + i;
        o[c] = (v[i] - mean) * rstd * G[c] + Bt[c];
    }
}

// ---------------- orchestration ----------------
extern "C" void kernel_launch(void* const* d_in, const int* in_sizes, int n_in,
                              void* d_out, int out_size, void* d_ws, size_t ws_size,
                              hipStream_t stream)
{
    const float* x       = (const float*)d_in[0];
    const float* lq_w    = (const float*)d_in[1];  const float* lq_b   = (const float*)d_in[2];
    const float* lk_w    = (const float*)d_in[3];  const float* lk_b   = (const float*)d_in[4];
    const float* lv_w    = (const float*)d_in[5];  const float* lv_b   = (const float*)d_in[6];
    const float* gq_w    = (const float*)d_in[7];  const float* gq_b   = (const float*)d_in[8];
    const float* gk_w    = (const float*)d_in[9];  const float* gk_b   = (const float*)d_in[10];
    const float* gv_w    = (const float*)d_in[11]; const float* gv_b   = (const float*)d_in[12];
    const float* h_out_w = (const float*)d_in[13]; const float* h_out_b= (const float*)d_in[14];
    const float* h_ln_g  = (const float*)d_in[15]; const float* h_ln_b = (const float*)d_in[16];
    const float* sq_w    = (const float*)d_in[17]; const float* sq_b   = (const float*)d_in[18];
    const float* sk_w    = (const float*)d_in[19]; const float* sk_b   = (const float*)d_in[20];
    const float* sv_w    = (const float*)d_in[21]; const float* sv_b   = (const float*)d_in[22];
    const float* s_out_w = (const float*)d_in[23]; const float* s_out_b= (const float*)d_in[24];
    const float* comb_w  = (const float*)d_in[25]; const float* comb_b = (const float*)d_in[26];
    const float* ln_g    = (const float*)d_in[27]; const float* ln_b   = (const float*)d_in[28];

    const int M = BB * SS;                 // 4096
    const size_t RD = (size_t)M * DD;      // 4M floats

    float* p = (float*)d_ws;
    float* lq    = p; p += RD;
    float* lk    = p; p += RD;
    float* lv    = p; p += RD;
    float* gq    = p; p += RD;
    float* gk    = p; p += (size_t)BB * GGC * DD;
    float* gv    = p; p += (size_t)BB * GGC * DD;
    float* sq    = p; p += RD;
    float* sk    = p; p += RD;
    float* sv    = p; p += RD;
    float* catLG = p; p += 2 * RD;
    float* hcomb = p; p += RD;
    float* sattn = p; p += RD;
    float* catHS = p; p += 2 * RD;
    float* comb  = p; p += RD;
    (void)ws_size; (void)in_sizes; (void)n_in; (void)out_size;

    dim3 blk(256);
    dim3 gFull(DD / 128, M / 128);         // 8 x 32
    dim3 gSmall(DD / 128, 1);              // gk/gv: M=64

    // Q/K/V projections (bf16 WMMA GEMMs)
    gemm_bias_wmma<<<gFull, blk, 0, stream>>>(x, DD, M, DD, lq_w, DD, lq_b, lq, DD, 0);
    gemm_bias_wmma<<<gFull, blk, 0, stream>>>(x, DD, M, DD, lk_w, DD, lk_b, lk, DD, 0);
    gemm_bias_wmma<<<gFull, blk, 0, stream>>>(x, DD, M, DD, lv_w, DD, lv_b, lv, DD, 0);
    gemm_bias_wmma<<<gFull, blk, 0, stream>>>(x, DD, M, DD, gq_w, DD, gq_b, gq, DD, 0);
    gemm_bias_wmma<<<gSmall, blk, 0, stream>>>(x, GST * DD, BB * GGC, DD, gk_w, DD, gk_b, gk, DD, 0);
    gemm_bias_wmma<<<gSmall, blk, 0, stream>>>(x, GST * DD, BB * GGC, DD, gv_w, DD, gv_b, gv, DD, 0);
    gemm_bias_wmma<<<gFull, blk, 0, stream>>>(x, DD, M, DD, sq_w, DD, sq_b, sq, DD, 0);
    gemm_bias_wmma<<<gFull, blk, 0, stream>>>(x, DD, M, DD, sk_w, DD, sk_b, sk, DD, 0);
    gemm_bias_wmma<<<gFull, blk, 0, stream>>>(x, DD, M, DD, sv_w, DD, sv_b, sv, DD, 0);

    // WMMA flash attention: local + global write into concat buffer [local | global]
    local_attn_wmma<<<BB * NWW * HH, 256, 0, stream>>>(lq, lk, lv, catLG, 2 * DD, 0);
    global_attn_wmma<<<BB * HH * (SS / 256), 256, 0, stream>>>(gq, gk, gv, catLG, 2 * DD, DD);

    // h_comb = concat @ h_out_w + b ; h_out = LN(x + h_comb) -> first half of catHS
    gemm_bias_wmma<<<gFull, blk, 0, stream>>>(catLG, 2 * DD, M, 2 * DD, h_out_w, DD, h_out_b, hcomb, DD, 0);
    add_ln_kernel<<<M, 256, 0, stream>>>(x, hcomb, h_ln_g, h_ln_b, catHS, 2 * DD, 0);

    // sliding-band WMMA attention, then s_out GEMM into second half of catHS
    sliding_attn_wmma<<<BB * HH * (SS / 256), 256, 0, stream>>>(sq, sk, sv, sattn);
    gemm_bias_wmma<<<gFull, blk, 0, stream>>>(sattn, DD, M, DD, s_out_w, DD, s_out_b, catHS, 2 * DD, DD);

    // combined = catHS @ comb_w + b ; out = LN(x + combined)
    gemm_bias_wmma<<<gFull, blk, 0, stream>>>(catHS, 2 * DD, M, 2 * DD, comb_w, DD, comb_b, comb, DD, 0);
    add_ln_kernel<<<M, 256, 0, stream>>>(x, comb, ln_g, ln_b, (float*)d_out, DD, 0);
}
